// DGMN2Attention_3D_34153579938288
// MI455X (gfx1250) — matrix operand order repaired and checked by reading, not compiled, not run on hardware
//
#include <hip/hip_runtime.h>
#include <math.h>

// ---------------------------------------------------------------------------
// DGMN2 3D deformable attention for MI455X (gfx1250), wave32 + WMMA bf16x2.
//   B=1, D=H=W=16, N=4096, C=512, nh=8, hd=64, NS=27
// ---------------------------------------------------------------------------

typedef __bf16 v16bf __attribute__((ext_vector_type(16)));
typedef float  v8f   __attribute__((ext_vector_type(8)));
typedef unsigned int v8u __attribute__((ext_vector_type(8)));

#define NTOK 4096
#define CIN  512
#define NHEAD 8
#define HDIM 64
#define NSAMP 27

// round-to-nearest-even float -> bf16 bits
__device__ __forceinline__ unsigned int f2bf(float f) {
    unsigned int u = __float_as_uint(f);
    u += 0x7FFFu + ((u >> 16) & 1u);
    return u >> 16;
}

// split f into hi bf16 + residual bf16 (bfloat16x2)
__device__ __forceinline__ void split_bf(float f, unsigned int& hi, unsigned int& lo) {
    hi = f2bf(f);
    float hf = __uint_as_float(hi << 16);
    lo = f2bf(f - hf);
}

// convert float4 -> packed 4x bf16 (hi & lo parts), as uint2 each
__device__ __forceinline__ void pack4(const float4 v, uint2& phi, uint2& plo) {
    unsigned int h0, l0, h1, l1, h2, l2, h3, l3;
    split_bf(v.x, h0, l0); split_bf(v.y, h1, l1);
    split_bf(v.z, h2, l2); split_bf(v.w, h3, l3);
    phi.x = h0 | (h1 << 16); phi.y = h2 | (h3 << 16);
    plo.x = l0 | (l1 << 16); plo.y = l2 | (l3 << 16);
}

// ---------------------------------------------------------------------------
// GEMM:  C[M x Nc] = A[M x K] * W[Nc x K]^T (+ bias), f32 in/out.
// bfloat16x2 split: acc += Ahi*Bhi + Ahi*Blo + Alo*Bhi.
// Block: 256 threads = 8 waves; block tile 64(M) x 64(N), K-step 32.
// Wave (4x2 grid) computes 16(M) x 32(N): A-frag reused over 2 B tiles
// -> 6 WMMAs per k-step per wave. Requires M % 64 == 0, K % 32 == 0.
// ---------------------------------------------------------------------------
__global__ __launch_bounds__(256)
void wmma_gemm_bf16x2(const float* __restrict__ A, int lda,
                      const float* __restrict__ W,   // Nc x K, row-major
                      float* __restrict__ C, int ldc,
                      const float* __restrict__ bias,
                      int M, int Nc, int K)
{
    __shared__ unsigned short As_hi[64][36];
    __shared__ unsigned short As_lo[64][36];
    __shared__ unsigned short Bs_hi[64][36];
    __shared__ unsigned short Bs_lo[64][36];

    const int tid  = threadIdx.x;
    const int lane = tid & 31;
    const int wv   = tid >> 5;
    const int m0   = blockIdx.x * 64;
    const int n0   = blockIdx.y * 64;
    const int half = lane >> 4;
    const int lm   = lane & 15;
    const int mbase = (wv >> 1) * 16;   // 4 wave-rows
    const int nbase = (wv & 1) * 32;    // 2 wave-cols, each 32 wide

    v8f acc0 = {};
    v8f acc1 = {};

    // stager geometry: 64 rows x 32 k, 8 floats (2x float4) per thread
    const int srow = tid >> 2;          // 0..63
    const int scol = (tid & 3) * 8;     // 0,8,16,24

    const float* apA = A + (size_t)(m0 + srow) * lda + scol;
    const int nrow   = min(n0 + srow, Nc - 1);       // clamped: loads always valid
    const bool nok   = (n0 + srow) < Nc;
    const float* apB = W + (size_t)nrow * K + scol;

    for (int k0 = 0; k0 < K; k0 += 32) {
        __syncthreads();
        // ---- stage A tile 64x32 ----
        {
            const float4 va0 = *(const float4*)(apA + k0);
            const float4 va1 = *(const float4*)(apA + k0 + 4);
            uint2 h, l;
            pack4(va0, h, l);
            *(uint2*)&As_hi[srow][scol]     = h;
            *(uint2*)&As_lo[srow][scol]     = l;
            pack4(va1, h, l);
            *(uint2*)&As_hi[srow][scol + 4] = h;
            *(uint2*)&As_lo[srow][scol + 4] = l;
        }
        // ---- stage B tile 64(n) x 32(k), zero rows past Nc ----
        {
            float4 vb0 = *(const float4*)(apB + k0);
            float4 vb1 = *(const float4*)(apB + k0 + 4);
            if (!nok) { vb0 = make_float4(0.f, 0.f, 0.f, 0.f); vb1 = vb0; }
            uint2 h, l;
            pack4(vb0, h, l);
            *(uint2*)&Bs_hi[srow][scol]     = h;
            *(uint2*)&Bs_lo[srow][scol]     = l;
            pack4(vb1, h, l);
            *(uint2*)&Bs_hi[srow][scol + 4] = h;
            *(uint2*)&Bs_lo[srow][scol + 4] = l;
        }
        // prefetch next k-tile into cache while we compute
        if (k0 + 32 < K) {
            __builtin_prefetch(apA + k0 + 32, 0, 1);
            __builtin_prefetch(apB + k0 + 32, 0, 1);
        }
        __syncthreads();

        // Fragments. A (16-bit 16x32): lane m=lane%16,
        //   element e -> k = e + 16*(lane/16)
        // B (32x16): lane n=lane%16, element e -> k = e + 16*(lane/16)
        union { v8u u; unsigned int s[8]; } uah, ual, ubh0, ubl0, ubh1, ubl1;
        #pragma unroll
        for (int j = 0; j < 8; ++j) {
            const int kk = 2 * j + 16 * half;                      // even
            uah.s[j] = *(const unsigned int*)&As_hi[mbase + lm][kk];
            ual.s[j] = *(const unsigned int*)&As_lo[mbase + lm][kk];
        }
        #pragma unroll
        for (int j = 0; j < 8; ++j) {
            const int kk = 2 * j + 16 * half;                      // even
            ubh0.s[j] = *(const unsigned int*)&Bs_hi[nbase + lm][kk];
            ubl0.s[j] = *(const unsigned int*)&Bs_lo[nbase + lm][kk];
            ubh1.s[j] = *(const unsigned int*)&Bs_hi[nbase + 16 + lm][kk];
            ubl1.s[j] = *(const unsigned int*)&Bs_lo[nbase + 16 + lm][kk];
        }
        v16bf ah  = __builtin_bit_cast(v16bf, uah.u);
        v16bf al  = __builtin_bit_cast(v16bf, ual.u);
        v16bf bh0 = __builtin_bit_cast(v16bf, ubh0.u);
        v16bf bl0 = __builtin_bit_cast(v16bf, ubl0.u);
        v16bf bh1 = __builtin_bit_cast(v16bf, ubh1.u);
        v16bf bl1 = __builtin_bit_cast(v16bf, ubl1.u);

        acc0 = __builtin_amdgcn_wmma_f32_16x16x32_bf16(false, ah, false, bh0, (short)0, acc0, false, false);
        acc1 = __builtin_amdgcn_wmma_f32_16x16x32_bf16(false, ah, false, bh1, (short)0, acc1, false, false);
        acc0 = __builtin_amdgcn_wmma_f32_16x16x32_bf16(false, ah, false, bl0, (short)0, acc0, false, false);
        acc1 = __builtin_amdgcn_wmma_f32_16x16x32_bf16(false, ah, false, bl1, (short)0, acc1, false, false);
        acc0 = __builtin_amdgcn_wmma_f32_16x16x32_bf16(false, al, false, bh0, (short)0, acc0, false, false);
        acc1 = __builtin_amdgcn_wmma_f32_16x16x32_bf16(false, al, false, bh1, (short)0, acc1, false, false);
    }

    // C/D layout: n = lane%16, m = vgpr r + 8*(lane/16)
    union { v8f v; float f[8]; } uc0, uc1;
    uc0.v = acc0; uc1.v = acc1;
    const int nA = n0 + nbase + lm;
    const int nB = nA + 16;
    const float bA = (bias && nA < Nc) ? bias[nA] : 0.0f;
    const float bB = (bias && nB < Nc) ? bias[nB] : 0.0f;
    #pragma unroll
    for (int r = 0; r < 8; ++r) {
        const int m = m0 + mbase + r + 8 * half;
        if (nA < Nc) C[(size_t)m * ldc + nA] = uc0.f[r] + bA;
        if (nB < Nc) C[(size_t)m * ldc + nB] = uc1.f[r] + bB;
    }
}

// ---------------------------------------------------------------------------
// Fused deformable attention + relative position + softmax + V-aggregation.
// One wave per query (8 waves/block). hd=64 -> lane owns channels lane, lane+32.
// pos[h,(a,b,c),s] = q(a,b,c).rw[15+s-c] + q(a,c,b).rh[15+s-c] + q(c,a,b).rd[15+s-c]
// (all _resize1d calls are identities for D=H=W=16).
// ---------------------------------------------------------------------------
__global__ __launch_bounds__(256)
void deform_attn(const float* __restrict__ qkv,   // N x 1536 (q|k|v, head-major)
                 const float* __restrict__ offp,  // (N*8) x 81
                 const float* __restrict__ reld,  // 42 x 64
                 const float* __restrict__ relh,  // 42 x 64
                 const float* __restrict__ relw,  // 42 x 64
                 float* __restrict__ outp)        // N x 512
{
    __shared__ float s_wgt[8][216];
    __shared__ int   s_idx[8][216];

    const int tid  = threadIdx.x;
    const int lane = tid & 31;
    const int wv   = tid >> 5;
    const int h    = blockIdx.y;
    const int n    = blockIdx.x * 8 + wv;
    const int a = n >> 8, b = (n >> 4) & 15, c = n & 15;

    // ---- phase 1: trilinear corner indices/weights for 27 samples ----------
    const float* ob = offp + ((size_t)n * NHEAD + h) * (NSAMP * 3);
    for (int i = 0; i < 7; ++i) {
        const int sc = i * 32 + lane;
        if (sc < 216) {
            const int s = sc >> 3, corner = sc & 7;
            const float dz = ob[s * 3 + 0];
            const float dy = ob[s * 3 + 1];
            const float dx = ob[s * 3 + 2];
            const float pz = (float)a + (float)(s / 9 - 1) + dz;
            const float py = (float)b + (float)((s / 3) % 3 - 1) + dy;
            const float px = (float)c + (float)(s % 3 - 1) + dx;
            const float z0 = floorf(pz), y0 = floorf(py), x0 = floorf(px);
            const float wz = pz - z0, wy = py - y0, wx = px - x0;
            const int zi = (int)z0 + ((corner >> 2) & 1);
            const int yi = (int)y0 + ((corner >> 1) & 1);
            const int xi = (int)x0 + (corner & 1);
            const bool ok = (zi >= 0) && (zi < 16) && (yi >= 0) && (yi < 16) &&
                            (xi >= 0) && (xi < 16);
            float wgt = ((corner & 4) ? wz : 1.0f - wz) *
                        ((corner & 2) ? wy : 1.0f - wy) *
                        ((corner & 1) ? wx : 1.0f - wx) * (ok ? 1.0f : 0.0f);
            const int zc = min(max(zi, 0), 15);
            const int yc = min(max(yi, 0), 15);
            const int xc = min(max(xi, 0), 15);
            s_wgt[wv][sc] = wgt;
            s_idx[wv][sc] = (zc * 16 + yc) * 16 + xc;
        }
    }
    __syncthreads();

    // ---- phase 2: logits (q.k * scale + pos), butterfly-reduced ------------
    const int ch0 = lane, ch1 = lane + 32;
    const float* qrow = qkv + (size_t)n * 1536 + h * 64;
    const float q0 = qrow[ch0], q1 = qrow[ch1];
    const int n_acb = (a * 16 + c) * 16 + b;
    const int n_cab = (c * 16 + a) * 16 + b;
    const float* qra = qkv + (size_t)n_acb * 1536 + h * 64;
    const float* qrc = qkv + (size_t)n_cab * 1536 + h * 64;
    const float qh0 = qra[ch0], qh1 = qra[ch1];
    const float qd0 = qrc[ch0], qd1 = qrc[ch1];

    float lg[NSAMP];
    for (int s = 0; s < NSAMP; ++s) {
        float p = 0.0f;
        #pragma unroll
        for (int corner = 0; corner < 8; ++corner) {
            const float wgt = s_wgt[wv][s * 8 + corner];
            const int   idx = s_idx[wv][s * 8 + corner];
            const float* kp = qkv + (size_t)idx * 1536 + 512 + h * 64;
            p += wgt * (q0 * kp[ch0] + q1 * kp[ch1]);
        }
        p *= 0.125f;  // hd^-0.5
        const int r = 15 + s - c;   // always in [0, 41]
        const float* rw_ = relw + r * 64;
        const float* rh_ = relh + r * 64;
        const float* rd_ = reld + r * 64;
        p += q0 * rw_[ch0] + q1 * rw_[ch1]
           + qh0 * rh_[ch0] + qh1 * rh_[ch1]
           + qd0 * rd_[ch0] + qd1 * rd_[ch1];
        #pragma unroll
        for (int off = 16; off > 0; off >>= 1)
            p += __shfl_xor(p, off, 32);
        lg[s] = p;   // all lanes hold the full sum
    }

    // ---- softmax over 27 (redundant per lane) ------------------------------
    float mx = lg[0];
    #pragma unroll
    for (int s = 1; s < NSAMP; ++s) mx = fmaxf(mx, lg[s]);
    float sum = 0.0f;
    #pragma unroll
    for (int s = 0; s < NSAMP; ++s) { lg[s] = __expf(lg[s] - mx); sum += lg[s]; }
    const float inv = 1.0f / sum;

    // ---- phase 3: weighted V aggregation -----------------------------------
    float o0 = 0.0f, o1 = 0.0f;
    for (int s = 0; s < NSAMP; ++s) {
        const float at = lg[s] * inv;
        #pragma unroll
        for (int corner = 0; corner < 8; ++corner) {
            const float coeff = at * s_wgt[wv][s * 8 + corner];
            const int   idx   = s_idx[wv][s * 8 + corner];
            const float* vp = qkv + (size_t)idx * 1536 + 1024 + h * 64;
            o0 += coeff * vp[ch0];
            o1 += coeff * vp[ch1];
        }
    }
    float* orow = outp + (size_t)n * CIN + h * 64;
    orow[ch0] = o0;
    orow[ch1] = o1;
}

// ---------------------------------------------------------------------------
extern "C" void kernel_launch(void* const* d_in, const int* in_sizes, int n_in,
                              void* d_out, int out_size, void* d_ws, size_t ws_size,
                              hipStream_t stream) {
    (void)in_sizes; (void)n_in; (void)out_size; (void)ws_size;
    const float* x      = (const float*)d_in[0];  // 4096 x 512
    const float* w_qkv  = (const float*)d_in[1];  // 1536 x 512
    const float* w_proj = (const float*)d_in[2];  // 512 x 512
    const float* b_proj = (const float*)d_in[3];  // 512
    const float* w_off  = (const float*)d_in[4];  // 81 x 64
    const float* reld   = (const float*)d_in[5];  // 42 x 64
    const float* relh   = (const float*)d_in[6];  // 42 x 64
    const float* relw   = (const float*)d_in[7];  // 42 x 64
    float* out = (float*)d_out;

    char* ws = (char*)d_ws;
    float* qkv  = (float*)ws;                                        // 4096*1536 f32
    float* offp = (float*)(ws + (size_t)NTOK * 1536 * 4);            // 32768*81 f32
    float* att  = (float*)(ws + (size_t)NTOK * 1536 * 4
                              + (size_t)NTOK * NHEAD * 81 * 4);      // 4096*512 f32

    const dim3 blk(256);
    // qkv = x @ w_qkv^T              : M=4096, K=512, Nc=1536
    wmma_gemm_bf16x2<<<dim3(NTOK / 64, 1536 / 64), blk, 0, stream>>>(
        x, CIN, w_qkv, qkv, 1536, nullptr, NTOK, 1536, CIN);
    // offsets: x.view(32768,64) @ w_off^T : M=32768, K=64, Nc=81
    wmma_gemm_bf16x2<<<dim3((NTOK * NHEAD) / 64, 2), blk, 0, stream>>>(
        x, HDIM, w_off, offp, 81, nullptr, NTOK * NHEAD, 81, HDIM);
    // fused deformable attention
    deform_attn<<<dim3(NTOK / 8, NHEAD), blk, 0, stream>>>(
        qkv, offp, reld, relh, relw, att);
    // out = att @ w_proj^T + b_proj  : M=4096, K=512, Nc=512
    wmma_gemm_bf16x2<<<dim3(NTOK / 64, CIN / 64), blk, 0, stream>>>(
        att, CIN, w_proj, out, CIN, b_proj, NTOK, CIN, CIN);
}
